// Attention_19533511262498
// MI455X (gfx1250) — matrix operand order repaired and checked by reading, not compiled, not run on hardware
//
#include <hip/hip_runtime.h>

// ---------------------------------------------------------------------------
// Types for CDNA5 WMMA (wave32): V_WMMA_F32_16X16X32_F16
// ---------------------------------------------------------------------------
typedef __attribute__((ext_vector_type(16))) _Float16 v16h;
typedef __attribute__((ext_vector_type(8)))  _Float16 v8h;
typedef __attribute__((ext_vector_type(8)))  float    v8f;
typedef unsigned int u32x4 __attribute__((ext_vector_type(4)));
typedef unsigned int u32x8 __attribute__((ext_vector_type(8)));

union VH { v16h v; v8h h2[2]; };

#define S_LEN    2048
#define DIM_Q    4096
#define DIM_KV   1024
#define N_HEADS  32
#define N_KV     8
#define HEAD_DIM 128
#define NEG_BIG  (-3.0e38f)

// ---------------------------------------------------------------------------
// f32 -> f16 cast (grid-stride)
// ---------------------------------------------------------------------------
__global__ void cast_f32_to_f16(const float* __restrict__ in,
                                _Float16* __restrict__ out, int n) {
    int i = blockIdx.x * blockDim.x + threadIdx.x;
    int stride = gridDim.x * blockDim.x;
    for (; i < n; i += stride) out[i] = (_Float16)in[i];
}

// ---------------------------------------------------------------------------
// RoPE (interleaved pairs) + cast to f16.  in: [S, nHeads, 128] f32
// ---------------------------------------------------------------------------
__global__ void rope_to_f16(const float* __restrict__ in,
                            const float* __restrict__ cosT,
                            const float* __restrict__ sinT,
                            _Float16* __restrict__ out, int nHeads) {
    int idx = blockIdx.x * blockDim.x + threadIdx.x;
    int total = S_LEN * nHeads * (HEAD_DIM / 2);
    if (idx >= total) return;
    int d2 = idx & 63;
    int t  = idx >> 6;
    int h  = t % nHeads;
    int s  = t / nHeads;
    size_t base = ((size_t)s * nHeads + h) * HEAD_DIM + 2 * d2;
    float xr = in[base], xi = in[base + 1];
    float c  = cosT[s * 64 + d2];
    float sn = sinT[s * 64 + d2];
    out[base]     = (_Float16)(xr * c - xi * sn);
    out[base + 1] = (_Float16)(xr * sn + xi * c);
}

// ---------------------------------------------------------------------------
// GEMM: C[M,N] (f32) = A[M,K] (f16, row-major) * W[N,K]^T (f16, row-major)
// 256 threads = 8 waves; wave grid 2(M) x 4(N); wave tile 32x64 (2x4 WMMA)
// -> block tile 64 x 256.  8 WMMAs per 6 fragment loads per K=32 step.
// A-frag (16x32 f16): lane m=L&15 holds row m; K chunks split by L>>4.
// B-frag (32x16 f16): lane n=L&15 holds col n = W row n; K half by L>>4.
// ---------------------------------------------------------------------------
__global__ __launch_bounds__(256)
void gemm_f16_wmma(const _Float16* __restrict__ A,
                   const _Float16* __restrict__ W,
                   float* __restrict__ C, int M, int N, int K) {
    const int tid  = threadIdx.x;
    const int wave = tid >> 5;
    const int lane = tid & 31;
    const int lm   = lane & 15;
    const int hi   = lane >> 4;
    const int wm   = wave & 1;
    const int wn   = wave >> 1;
    const int m0   = blockIdx.x * 64  + wm * 32;
    const int n0   = blockIdx.y * 256 + wn * 64;

    v8f acc[2][4];
#pragma unroll
    for (int a = 0; a < 2; ++a)
#pragma unroll
        for (int b = 0; b < 4; ++b)
            acc[a][b] = (v8f){0.f,0.f,0.f,0.f,0.f,0.f,0.f,0.f};

    for (int k = 0; k < K; k += 32) {
        VH af[2], bf[4];
#pragma unroll
        for (int tm = 0; tm < 2; ++tm) {
            const _Float16* ap = A + (size_t)(m0 + tm * 16 + lm) * K + k;
            __builtin_prefetch(ap + 64, 0, 1);
            af[tm].h2[0] = *(const v8h*)(ap + 8 * hi);
            af[tm].h2[1] = *(const v8h*)(ap + 16 + 8 * hi);
        }
#pragma unroll
        for (int tn = 0; tn < 4; ++tn) {
            const _Float16* wp = W + (size_t)(n0 + tn * 16 + lm) * K + k + 8 * hi;
            __builtin_prefetch(wp + 64, 0, 1);
            bf[tn].h2[0] = *(const v8h*)(wp);
            bf[tn].h2[1] = *(const v8h*)(wp + 16);
        }
#pragma unroll
        for (int tm = 0; tm < 2; ++tm)
#pragma unroll
            for (int tn = 0; tn < 4; ++tn)
                acc[tm][tn] = __builtin_amdgcn_wmma_f32_16x16x32_f16(
                    false, af[tm].v, false, bf[tn].v,
                    (short)0, acc[tm][tn], false, false);
    }

#pragma unroll
    for (int tm = 0; tm < 2; ++tm)
#pragma unroll
        for (int tn = 0; tn < 4; ++tn)
#pragma unroll
            for (int i = 0; i < 8; ++i) {
                int row = m0 + tm * 16 + i + 8 * hi;
                int col = n0 + tn * 16 + lm;
                C[(size_t)row * N + col] = acc[tm][tn][i];
            }
}

// ---------------------------------------------------------------------------
// Flash attention (causal, GQA 4:1).  Block = (q-block of 64 rows, head).
// V tile is DMA'd into LDS by the Tensor Data Mover (tensor_load_to_lds),
// then transposed LDS->LDS into B-fragment-friendly layout.
// Online softmax in C-fragment layout with width-16 butterfly shuffles;
// P goes through wave-private LDS to reach the A-fragment layout.
// ---------------------------------------------------------------------------
__global__ __launch_bounds__(128)
void attn_wmma(const _Float16* __restrict__ q16,
               const _Float16* __restrict__ k16,
               const _Float16* __restrict__ v16,
               _Float16* __restrict__ o16) {
    __shared__ alignas(16) _Float16 VLds[64][HEAD_DIM];  // raw V tile (TDM dest)
    __shared__ alignas(16) _Float16 VtLds[HEAD_DIM][72]; // V^T tile [d][s]
    __shared__ alignas(16) _Float16 pLds[4][16][72];     // per-wave P staging

    const int h    = blockIdx.y;
    const int hk   = h >> 2;
    const int qb   = blockIdx.x;
    const int tid  = threadIdx.x;
    const int wave = tid >> 5;
    const int lane = tid & 31;
    const int lm   = lane & 15;
    const int hi   = lane >> 4;
    const int m0   = qb * 64 + wave * 16;
    const float scale = 0.08838834764831845f; // 1/sqrt(128)

    // Q fragments for this wave's 16 rows: 4 chunks of K=32 (d = 0..127)
    VH qf[4];
    {
        const _Float16* qrow = q16 + (size_t)(m0 + lm) * DIM_Q + h * HEAD_DIM;
#pragma unroll
        for (int kc = 0; kc < 4; ++kc) {
            qf[kc].h2[0] = *(const v8h*)(qrow + kc * 32 + 8 * hi);
            qf[kc].h2[1] = *(const v8h*)(qrow + kc * 32 + 16 + 8 * hi);
        }
    }

    v8f acc[8];
#pragma unroll
    for (int j = 0; j < 8; ++j)
        acc[j] = (v8f){0.f,0.f,0.f,0.f,0.f,0.f,0.f,0.f};
    float rowmax[8], rowsum[8];
#pragma unroll
    for (int i = 0; i < 8; ++i) { rowmax[i] = NEG_BIG; rowsum[i] = 0.f; }

    const unsigned ldsV = (unsigned)(uintptr_t)(&VLds[0][0]); // LDS byte offset

    for (int kb = 0; kb <= qb; ++kb) {
        const int s0 = kb * 64;

        // ---- TDM: DMA V block [64 rows x 128 halves] into VLds (row-major)
        __syncthreads();                    // WAR: prior readers of VLds/VtLds
        if (wave == 0) {
            unsigned long long ga = (unsigned long long)(uintptr_t)(
                v16 + (size_t)s0 * DIM_KV + hk * HEAD_DIM);
            u32x4 g0;
            g0[0] = 1u;                                   // count=1, user desc
            g0[1] = ldsV;                                 // lds_addr
            g0[2] = (unsigned)(ga & 0xffffffffu);         // global_addr[31:0]
            g0[3] = (unsigned)((ga >> 32) & 0x01ffffffu)  // global_addr[56:32]
                    | (2u << 30);                         // type = 2 (image)
            u32x8 g1;
            g1[0] = 0x00010000u;          // workgroup_mask=0, data_size=2B
            g1[1] = (unsigned)HEAD_DIM << 16;  // tensor_dim0 = 128 (low bits)
            g1[2] = 64u << 16;                 // tensor_dim1 = 64 (low bits)
            g1[3] = (unsigned)HEAD_DIM << 16;  // tile_dim0 = 128
            g1[4] = 64u;                       // tile_dim1 = 64, tile_dim2 = 0
            g1[5] = (unsigned)DIM_KV;          // tensor_dim0_stride = 1024
            g1[6] = 0u;
            g1[7] = 0u;
            asm volatile("tensor_load_to_lds %0, %1"
                         :: "s"(g0), "s"(g1)
                         : "memory");
            __builtin_amdgcn_s_wait_tensorcnt(0);
        }
        __syncthreads();

        // ---- transpose LDS->LDS: VtLds[d][s] = VLds[s][d]
        for (int it = 0; it < 8; ++it) {
            int flat = (it * 128 + tid) * 8;          // element index in 64x128
            int srow = flat >> 7;
            int d    = flat & 127;
            v8h vv = *(const v8h*)&VLds[srow][d];
#pragma unroll
            for (int e = 0; e < 8; ++e) VtLds[d + e][srow] = vv[e];
        }
        __syncthreads();

        // ---- S = Q K^T  (16 x 64, as 4 WMMA column tiles)
        v8f sc[4];
#pragma unroll
        for (int t = 0; t < 4; ++t) {
            sc[t] = (v8f){0.f,0.f,0.f,0.f,0.f,0.f,0.f,0.f};
#pragma unroll
            for (int kc = 0; kc < 4; ++kc) {
                VH bf;
                const _Float16* kp = k16 +
                    (size_t)(s0 + t * 16 + lm) * DIM_KV + hk * HEAD_DIM +
                    kc * 32 + 8 * hi;
                bf.h2[0] = *(const v8h*)(kp);
                bf.h2[1] = *(const v8h*)(kp + 16);
                sc[t] = __builtin_amdgcn_wmma_f32_16x16x32_f16(
                    false, qf[kc].v, false, bf.v, (short)0, sc[t], false, false);
            }
        }

        // ---- scale + causal mask (diagonal block only has masked entries)
#pragma unroll
        for (int t = 0; t < 4; ++t)
#pragma unroll
            for (int i = 0; i < 8; ++i) {
                float s = sc[t][i] * scale;
                int row = m0 + i + 8 * hi;
                int col = s0 + t * 16 + lm;
                sc[t][i] = (col <= row) ? s : NEG_BIG;
            }

        // ---- online softmax: per-row max over 64 cols (width-16 butterflies)
        float corr[8], mloc[8];
#pragma unroll
        for (int i = 0; i < 8; ++i) {
            float v = fmaxf(fmaxf(sc[0][i], sc[1][i]),
                            fmaxf(sc[2][i], sc[3][i]));
#pragma unroll
            for (int off = 8; off >= 1; off >>= 1)
                v = fmaxf(v, __shfl_xor(v, off, 16));
            float mnew = fmaxf(rowmax[i], v);
            corr[i]    = __expf(rowmax[i] - mnew);
            rowmax[i]  = mnew;
            mloc[i]    = mnew;
        }
#pragma unroll
        for (int i = 0; i < 8; ++i) {
            float sm = 0.f;
#pragma unroll
            for (int t = 0; t < 4; ++t) {
                float p = __expf(sc[t][i] - mloc[i]);
                sc[t][i] = p;
                sm += p;
            }
#pragma unroll
            for (int off = 8; off >= 1; off >>= 1)
                sm += __shfl_xor(sm, off, 16);
            rowsum[i] = rowsum[i] * corr[i] + sm;
        }
#pragma unroll
        for (int j = 0; j < 8; ++j)
#pragma unroll
            for (int i = 0; i < 8; ++i)
                acc[j][i] *= corr[i];

        // ---- P: C-layout -> LDS -> A-fragment layout (wave-private)
#pragma unroll
        for (int t = 0; t < 4; ++t)
#pragma unroll
            for (int i = 0; i < 8; ++i)
                pLds[wave][i + 8 * hi][t * 16 + lm] = (_Float16)sc[t][i];

        VH pf[2];
#pragma unroll
        for (int kc = 0; kc < 2; ++kc) {
            const _Float16* pr = &pLds[wave][lm][kc * 32];
            pf[kc].h2[0] = *(const v8h*)(pr + 8 * hi);
            pf[kc].h2[1] = *(const v8h*)(pr + 16 + 8 * hi);
        }

        // ---- O += P V : B-frags are contiguous rows of VtLds
#pragma unroll
        for (int j = 0; j < 8; ++j) {
#pragma unroll
            for (int kc = 0; kc < 2; ++kc) {
                VH bf;
                const _Float16* vp = &VtLds[j * 16 + lm][kc * 32 + 8 * hi];
                bf.h2[0] = *(const v8h*)(vp);
                bf.h2[1] = *(const v8h*)(vp + 16);
                acc[j] = __builtin_amdgcn_wmma_f32_16x16x32_f16(
                    false, pf[kc].v, false, bf.v, (short)0, acc[j], false, false);
            }
        }
    }

    // ---- normalize and emit f16 attention output [S, 4096]
#pragma unroll
    for (int i = 0; i < 8; ++i) rowsum[i] = 1.0f / rowsum[i];
#pragma unroll
    for (int j = 0; j < 8; ++j)
#pragma unroll
        for (int i = 0; i < 8; ++i) {
            int row = m0 + i + 8 * hi;
            int col = h * HEAD_DIM + j * 16 + lm;
            o16[(size_t)row * DIM_Q + col] = (_Float16)(acc[j][i] * rowsum[i]);
        }
}

// ---------------------------------------------------------------------------
// Host launcher
// ---------------------------------------------------------------------------
extern "C" void kernel_launch(void* const* d_in, const int* in_sizes, int n_in,
                              void* d_out, int out_size, void* d_ws, size_t ws_size,
                              hipStream_t stream) {
    const float* x    = (const float*)d_in[0];
    const float* wq   = (const float*)d_in[1];
    const float* wk   = (const float*)d_in[2];
    const float* wv   = (const float*)d_in[3];
    const float* wo   = (const float*)d_in[4];
    const float* fcos = (const float*)d_in[5];
    const float* fsin = (const float*)d_in[6];
    float* out = (float*)d_out;

    char* p = (char*)d_ws;
    _Float16* x16  = (_Float16*)(p);
    _Float16* wq16 = (_Float16*)(p + (16ull  << 20));
    _Float16* wk16 = (_Float16*)(p + (48ull  << 20));
    _Float16* wv16 = (_Float16*)(p + (56ull  << 20));
    _Float16* wo16 = (_Float16*)(p + (64ull  << 20));
    float*    qf   = (float*)   (p + (96ull  << 20));
    float*    kf   = (float*)   (p + (128ull << 20));
    float*    vf   = (float*)   (p + (136ull << 20));
    _Float16* q16  = (_Float16*)(p + (144ull << 20));
    _Float16* k16  = (_Float16*)(p + (160ull << 20));
    _Float16* v16  = (_Float16*)(p + (164ull << 20));
    _Float16* a16  = (_Float16*)(p + (168ull << 20));

    const int nx  = S_LEN * DIM_Q;     // 8388608
    const int nwq = DIM_Q * DIM_Q;     // 16777216
    const int nwk = DIM_KV * DIM_Q;    // 4194304
    const int nkv = S_LEN * DIM_KV;    // 2097152

    // 1) weight/activation down-conversion to f16
    cast_f32_to_f16<<<4096, 256, 0, stream>>>(x,  x16,  nx);
    cast_f32_to_f16<<<4096, 256, 0, stream>>>(wq, wq16, nwq);
    cast_f32_to_f16<<<4096, 256, 0, stream>>>(wk, wk16, nwk);
    cast_f32_to_f16<<<4096, 256, 0, stream>>>(wv, wv16, nwk);
    cast_f32_to_f16<<<4096, 256, 0, stream>>>(wo, wo16, nwq);

    // 2) Q/K/V projections (block tile 64 x 256)
    gemm_f16_wmma<<<dim3(S_LEN / 64, DIM_Q / 256),  256, 0, stream>>>(
        x16, wq16, qf, S_LEN, DIM_Q, DIM_Q);
    gemm_f16_wmma<<<dim3(S_LEN / 64, DIM_KV / 256), 256, 0, stream>>>(
        x16, wk16, kf, S_LEN, DIM_KV, DIM_Q);
    gemm_f16_wmma<<<dim3(S_LEN / 64, DIM_KV / 256), 256, 0, stream>>>(
        x16, wv16, vf, S_LEN, DIM_KV, DIM_Q);

    // 3) RoPE on Q/K (+cast), plain cast for V
    rope_to_f16<<<(S_LEN * N_HEADS * 64 + 255) / 256, 256, 0, stream>>>(
        qf, fcos, fsin, q16, N_HEADS);
    rope_to_f16<<<(S_LEN * N_KV * 64 + 255) / 256, 256, 0, stream>>>(
        kf, fcos, fsin, k16, N_KV);
    cast_f32_to_f16<<<2048, 256, 0, stream>>>(vf, v16, nkv);

    // 4) causal flash attention (TDM-staged V)
    attn_wmma<<<dim3(S_LEN / 64, N_HEADS), 128, 0, stream>>>(q16, k16, v16, a16);

    // 5) output projection -> f32 d_out
    gemm_f16_wmma<<<dim3(S_LEN / 64, DIM_Q / 256), 256, 0, stream>>>(
        a16, wo16, out, S_LEN, DIM_Q, DIM_Q);
}
